// unit_vit_3624952398503
// MI455X (gfx1250) — matrix operand order repaired and checked by reading, not compiled, not run on hardware
//
#include <hip/hip_runtime.h>
#include <hip/hip_bf16.h>

typedef _Float16 v16h __attribute__((ext_vector_type(16)));
typedef float    v8f  __attribute__((ext_vector_type(8)));
typedef unsigned int u32x4 __attribute__((ext_vector_type(4)));
typedef int          i32x8 __attribute__((ext_vector_type(8)));
typedef int          i32x4 __attribute__((ext_vector_type(4)));

#define TVX   3000      // T*V
#define NCOLS 96000     // N*T*V
#define CIN_  96
#define DIM_  192
#define NB_   32
#define TT_   120
#define VJ_   25
#define HH_   6
#define HD_   32
#define GG_   5
#define ECOLS 19200     // N*T*G
#define SCALE_ 0.17677669529663687f   // 32^-0.5

#if defined(__HIP_DEVICE_COMPILE__) && __has_builtin(__builtin_amdgcn_tensor_load_to_lds) && \
    __has_builtin(__builtin_amdgcn_s_wait_tensorcnt)
#define HAVE_TDM 1
#else
#define HAVE_TDM 0
#endif
#if defined(__HIP_DEVICE_COMPILE__) && !HAVE_TDM
#warning "CDNA5 TDM builtin not available on this toolchain; using plain LDS staging fallback"
#endif

// ---------------- packed fragment-order layouts (CDNA5 wave32 WMMA) ----------------
// X/B-matrix order: lane L of col-tile nt reads its v16h fragment at [(nt*32+lane)*16].
__device__ __forceinline__ int frag_off_x(int kk, int j) {
  return ((j >> 4) * 32 + (j & 15) + ((kk & 16) ? 16 : 0)) * 16 + (kk & 15);
}
// W/A-matrix order: lane reads its fragment at [m*32 + 16*(lane>=16)].
__device__ __forceinline__ int frag_off_w(int m, int kk) {
  return m * 32 + (kk & 31);
}
// global packed index for a [K rows][cols] f16 tensor stored as (colTile, kTile) 2048-half tiles
__device__ __forceinline__ size_t packed_idx(int ktCount, int row, int col) {
  return ((size_t)(col >> 6) * ktCount + (row >> 5)) * 2048 + frag_off_x(row & 31, col & 63);
}

__device__ __forceinline__ v8f wmma16(const v16h& a, const v16h& b, const v8f& c) {
  return __builtin_amdgcn_wmma_f32_16x16x32_f16(false, a, false, b, (short)0, c, false, false);
}

// ---------------- TDM: one descriptor copies a whole tile into LDS ----------------
#if HAVE_TDM
__device__ __forceinline__ void tdm_copy_f16(const _Float16* gsrc, _Float16* ldst, int elems) {
  unsigned long long ga = (unsigned long long)gsrc;
  unsigned int la = (unsigned int)(unsigned long long)ldst;   // low 32 bits = LDS byte address
  u32x4 g0;
  g0[0] = 1u;                                                  // count=1 (valid), no gather
  g0[1] = la;                                                  // lds_addr
  g0[2] = (unsigned int)(ga & 0xFFFFFFFFull);                  // global_addr[31:0]
  g0[3] = (unsigned int)((ga >> 32) & 0x01FFFFFFull) | 0x80000000u; // addr[56:32] | type=2
  i32x8 g1;
  g1[0] = 0x10000;                                             // data_size = 2 bytes
  g1[1] = (int)((unsigned)elems << 16);                        // tensor_dim0[15:0]
  g1[2] = (int)(((unsigned)elems >> 16) | (1u << 16));         // tensor_dim0[31:16], tensor_dim1=1
  g1[3] = (int)((unsigned)elems << 16);                        // tile_dim0 = elems
  g1[4] = 1;                                                   // tile_dim1=1, tile_dim2 unused
  g1[5] = elems;                                               // tensor_dim0_stride
  g1[6] = 0; g1[7] = 0;
  i32x4 z4 = {0, 0, 0, 0};
#if __clang_major__ >= 23
  i32x8 z8 = {0, 0, 0, 0, 0, 0, 0, 0};
  __builtin_amdgcn_tensor_load_to_lds(g0, g1, z4, z4, z8, 0);
#else
  __builtin_amdgcn_tensor_load_to_lds(g0, g1, z4, z4, 0);
#endif
}
#endif

// ---------------- weight packing (f32 -> f16 fragment order, done once) ----------------
__global__ __launch_bounds__(256) void pack_w_kernel(const float* __restrict__ src,
                                                     _Float16* __restrict__ dst,
                                                     int Mtot, int K, int mOff) {
  int i = blockIdx.x * 256 + threadIdx.x;
  if (i >= Mtot * K) return;
  int m = i / K, kg = i - m * K;
  int mg = m + mOff;
  dst[((size_t)(mg >> 6) * (K >> 5) + (kg >> 5)) * 2048 + frag_off_w(mg & 63, kg & 31)] =
      (_Float16)src[i];
}
__global__ __launch_bounds__(256) void pack_w_gcn_kernel(const float* __restrict__ src,
                                                         _Float16* __restrict__ dst) {
  int i = blockIdx.x * 256 + threadIdx.x;
  if (i >= DIM_ * 3 * CIN_) return;
  int m = i / (3 * CIN_), kg = i - m * (3 * CIN_);
  int kp = kg / CIN_, c = kg - kp * CIN_;
  float v = src[((size_t)kp * DIM_ + m) * CIN_ + c];   // gcn_w[k][o][c] -> W[o][k*96+c]
  dst[((size_t)(m >> 6) * 9 + (kg >> 5)) * 2048 + frag_off_w(m & 63, kg & 31)] = (_Float16)v;
}
__global__ __launch_bounds__(256) void pack_w_proj_kernel(const float* __restrict__ src,
                                                          _Float16* __restrict__ dst) {
  int i = blockIdx.x * 256 + threadIdx.x;
  if (i >= HH_ * HD_ * HD_) return;
  int h = i >> 10, r = i & 1023;
  dst[h * 1024 + frag_off_w(r >> 5, r & 31)] = (_Float16)src[i];
}

// ---------------- LayerNorm over channels -> packed XN ----------------
__global__ __launch_bounds__(128) void layernorm_kernel(const float* __restrict__ x,
                                                        const float* __restrict__ g,
                                                        const float* __restrict__ b,
                                                        _Float16* __restrict__ xnp) {
  int col = blockIdx.x * blockDim.x + threadIdx.x;
  if (col >= NCOLS) return;
  int n = col / TVX, tv = col - n * TVX;
  const float* xp = x + (size_t)n * CIN_ * TVX + tv;
  float s = 0.f, s2 = 0.f;
  for (int c = 0; c < CIN_; ++c) { float v = xp[(size_t)c * TVX]; s += v; s2 += v * v; }
  float mu  = s * (1.0f / CIN_);
  float var = s2 * (1.0f / CIN_) - mu * mu;
  float inv = rsqrtf(var + 1e-5f);
  for (int c = 0; c < CIN_; ++c)
    xnp[packed_idx(3, c, col)] = (_Float16)((xp[(size_t)c * TVX] - mu) * inv * g[c] + b[c]);
}

// ---------------- group-mean pooling -> packed Z ----------------
__global__ __launch_bounds__(128) void zpool_kernel(const float* __restrict__ x,
                                                    const int* __restrict__ jl,
                                                    _Float16* __restrict__ zp) {
  int idx = blockIdx.x * blockDim.x + threadIdx.x;
  if (idx >= CIN_ * NB_ * TT_) return;
  int c = idx / (NB_ * TT_);
  int nt = idx - c * (NB_ * TT_);
  int n = nt / TT_, t = nt - n * TT_;
  const float* xp = x + (((size_t)n * CIN_ + c) * TT_ + t) * VJ_;
  float acc[GG_] = {0, 0, 0, 0, 0};
  float cnt[GG_] = {0, 0, 0, 0, 0};
  for (int v = 0; v < VJ_; ++v) { int g = jl[v]; acc[g] += xp[v]; cnt[g] += 1.0f; }
  int colE = n * (TT_ * GG_) + t * GG_;
  for (int g = 0; g < GG_; ++g)
    zp[packed_idx(3, c, colE + g)] = (_Float16)(acc[g] / cnt[g]);
}

// ---------------- generic WMMA GEMM: OUT[M][cols] = W[M][K] * X[K][cols] ----------------
// IN_MODE: 0 = f32 x-layout strided (convert+scatter), 1 = packed f16 tiles (TDM copy)
// OUT    : 0 = f32 flat, 1 = f16 flat, 2 = d_out write, 3 = gcn epilogue (+= relu(bn(...)))
template <int IN_MODE, int OUT_MODE>
__global__ __launch_bounds__(128) void gemm_wmma(const void* __restrict__ Xin,
                                                 const _Float16* __restrict__ Wp,
                                                 void* __restrict__ Out, int K, int ncols,
                                                 const float* __restrict__ e0,
                                                 const float* __restrict__ e1,
                                                 const float* __restrict__ e2) {
  __shared__ alignas(32) _Float16 ldsX[2048];
  __shared__ alignas(32) _Float16 ldsW[2048];
  const int tid  = threadIdx.x;
  const int wave = tid >> 5, lane = tid & 31;
  const int colBase = blockIdx.x * 64;
  const int rowBase = blockIdx.y * 64;
  const int Kt = K >> 5;
  v8f acc[4] = {};
  for (int kt = 0; kt < Kt; ++kt) {
    const _Float16* wsrc = Wp + ((size_t)blockIdx.y * Kt + kt) * 2048;
    if (IN_MODE == 0) {
      const int k0 = kt << 5;
      for (int e = tid; e < 2048; e += 128) {
        int kk = e >> 6, j = e & 63;
        int col = colBase + j;
        int n = col / TVX;
        float xv = ((const float*)Xin)[((size_t)n * K + (k0 + kk)) * TVX + (col - n * TVX)];
        ldsX[frag_off_x(kk, j)] = (_Float16)xv;
      }
#if HAVE_TDM
      if (wave == 0) { tdm_copy_f16(wsrc, ldsW, 2048); __builtin_amdgcn_s_wait_tensorcnt(0); }
#else
      for (int e = tid; e < 2048; e += 128) ldsW[e] = wsrc[e];
#endif
    } else {
      const _Float16* xsrc = (const _Float16*)Xin + ((size_t)blockIdx.x * Kt + kt) * 2048;
#if HAVE_TDM
      if (wave == 0) { tdm_copy_f16(xsrc, ldsX, 2048); __builtin_amdgcn_s_wait_tensorcnt(0); }
      else if (wave == 1) { tdm_copy_f16(wsrc, ldsW, 2048); __builtin_amdgcn_s_wait_tensorcnt(0); }
#else
      for (int e = tid; e < 2048; e += 128) { ldsX[e] = xsrc[e]; ldsW[e] = wsrc[e]; }
#endif
    }
    __syncthreads();
    v16h a = *(const v16h*)(ldsW + (wave * 16 + (lane & 15)) * 32 + ((lane & 16) ? 16 : 0));
#pragma unroll
    for (int nt = 0; nt < 4; ++nt) {
      v16h b = *(const v16h*)(ldsX + (nt * 32 + lane) * 16);
      acc[nt] = wmma16(a, b, acc[nt]);
    }
    __syncthreads();
  }
  const int mrow = rowBase + wave * 16 + ((lane & 16) ? 8 : 0);
  const int nc = lane & 15;
#pragma unroll
  for (int nt = 0; nt < 4; ++nt) {
    int col = colBase + nt * 16 + nc;
#pragma unroll
    for (int r = 0; r < 8; ++r) {
      int m = mrow + r;
      float v = acc[nt][r];
      if (OUT_MODE == 0) {
        ((float*)Out)[(size_t)m * ncols + col] = v;
      } else if (OUT_MODE == 1) {
        ((_Float16*)Out)[(size_t)m * ncols + col] = (_Float16)v;
      } else if (OUT_MODE == 2) {
        int n = col / TVX;
        ((float*)Out)[((size_t)n * DIM_ + m) * TVX + (col - n * TVX)] = v;
      } else {
        float bsum = e0[m] + e0[DIM_ + m] + e0[2 * DIM_ + m];
        float val = (v + bsum) * rsqrtf(1.0f + 1e-5f) * e1[m] + e2[m];
        val = fmaxf(val, 0.0f);
        int n = col / TVX;
        ((float*)Out)[((size_t)n * DIM_ + m) * TVX + (col - n * TVX)] += val;
      }
    }
  }
}

// ---------------- GCN aggregation via WMMA: agg_k = X[96x25] * (A.M)_k[25x25] ----------------
__global__ __launch_bounds__(128) void agg_kernel(const float* __restrict__ x,
                                                  const float* __restrict__ A,
                                                  const float* __restrict__ Mm,
                                                  _Float16* __restrict__ aggp) {
  __shared__ alignas(32) _Float16 fX[CIN_ * 32];    // A-order: rows c(96), k = v (25 pad 32)
  __shared__ alignas(32) _Float16 fAM[3 * 1024];    // B-order per k: k = v(25 pad 32), n = w
  const int tid = threadIdx.x;
  const int wave = tid >> 5, lane = tid & 31;
  const int nt = blockIdx.x;
  const int n = nt / TT_, t = nt - n * TT_;
  for (int i = tid; i < CIN_ * 32; i += 128) fX[i] = (_Float16)0.f;      // zero K padding
  for (int i = tid; i < 3 * 1024; i += 128) fAM[i] = (_Float16)0.f;
  __syncthreads();
  for (int i = tid; i < CIN_ * VJ_; i += 128) {
    int c = i / VJ_, v = i - c * VJ_;
    fX[frag_off_w(c, v)] = (_Float16)x[(((size_t)n * CIN_ + c) * TT_ + t) * VJ_ + v];
  }
  for (int i = tid; i < 3 * VJ_ * VJ_; i += 128) {
    int k = i / (VJ_ * VJ_), r = i - k * (VJ_ * VJ_);
    int v = r / VJ_, w = r - v * VJ_;
    fAM[k * 1024 + frag_off_x(v, w)] = (_Float16)(A[i] * Mm[i]);
  }
  __syncthreads();
  const int colbase = n * TVX + t * VJ_;
  for (int idx = wave; idx < 36; idx += 4) {
    int k = idx / 12, rem = idx - k * 12, rt = rem >> 1, ct = rem & 1;
    v16h a = *(const v16h*)(fX + (rt * 16 + (lane & 15)) * 32 + ((lane & 16) ? 16 : 0));
    v16h b = *(const v16h*)(fAM + k * 1024 + (ct * 32 + lane) * 16);
    v8f c0 = {};
    c0 = wmma16(a, b, c0);
    int w = ct * 16 + (lane & 15);
    if (w < VJ_) {
#pragma unroll
      for (int r = 0; r < 8; ++r) {
        int c = rt * 16 + ((lane & 16) ? 8 : 0) + r;
        aggp[packed_idx(9, k * CIN_ + c, colbase + w)] = (_Float16)c0[r];
      }
    }
  }
}

// ---------------- attention via WMMA (per (n,t,h) block) ----------------
// VKQ rows: V = 0.., K1 = 192.., K2 = 384.., Q1 = 576.., Q2 = 768..
__global__ __launch_bounds__(128) void attn_kernel(
    const _Float16* __restrict__ VKQ, const float* __restrict__ Ebuf,
    const float* __restrict__ rpe, const float* __restrict__ w1,
    const float* __restrict__ outer_w, const int* __restrict__ hops,
    const int* __restrict__ jl, const float* __restrict__ alpha,
    const float* __restrict__ lq1, const float* __restrict__ lk1,
    const float* __restrict__ lq2, const float* __restrict__ lk2,
    _Float16* __restrict__ xsp) {
  __shared__ alignas(32) _Float16 fQ1[1024], fQ2[1024];    // A-order: rows nj, k = c
  __shared__ alignas(32) _Float16 fKE1[1024], fKE2[1024];  // B-order: k = c, n = m (K+Ek)
  __shared__ alignas(32) _Float16 fRpe[1024];              // B-order: k = c, n = hop
  __shared__ alignas(32) _Float16 fVv[1024];               // B-order: k = m (pad 32), n = c
  __shared__ alignas(32) _Float16 fWc[1024];               // A-order: rows nj, k = m (pad 32)
  __shared__ float sMain1[1024], sQr1[1024], sMain2[1024], sQr2[1024];  // 32x32 C tiles
  __shared__ float sEk[VJ_ * HD_];
  __shared__ float sD[32], sW1[HD_];
  __shared__ float sOut[VJ_ * VJ_];
  __shared__ int   sHops[VJ_ * VJ_];
  __shared__ float sLam, sAlpha;
  const int tid = threadIdx.x;
  const int wave = tid >> 5, lane = tid & 31;
  const int ntid = blockIdx.x;
  const int h = blockIdx.y;
  const int n = ntid / TT_, t = ntid - n * TT_;
  const size_t colbase = (size_t)n * TVX + t * VJ_;
  const int ecol = n * (TT_ * GG_) + t * GG_;
  // zero K-padded operands (avoid NaN from 0*garbage)
  for (int i = tid; i < 1024; i += 128) { fVv[i] = (_Float16)0.f; fWc[i] = (_Float16)0.f; }
  __syncthreads();
  for (int i = tid; i < VJ_ * HD_; i += 128) {
    int m = i >> 5, c = i & 31;
    int r = h * HD_ + c;
    float vv = (float)VKQ[(size_t)r * NCOLS + colbase + m];
    float k1 = (float)VKQ[(size_t)(192 + r) * NCOLS + colbase + m];
    float k2 = (float)VKQ[(size_t)(384 + r) * NCOLS + colbase + m];
    float q1 = (float)VKQ[(size_t)(576 + r) * NCOLS + colbase + m];
    float q2 = (float)VKQ[(size_t)(768 + r) * NCOLS + colbase + m];
    float ek = Ebuf[(size_t)r * ECOLS + ecol + jl[m]];
    sEk[i] = ek;
    fQ1[frag_off_w(m, c)] = (_Float16)q1;
    fQ2[frag_off_w(m, c)] = (_Float16)q2;
    fKE1[frag_off_x(c, m)] = (_Float16)(k1 + ek);
    fKE2[frag_off_x(c, m)] = (_Float16)(k2 + ek);
    fVv[frag_off_x(m, c)] = (_Float16)vv;
    fRpe[frag_off_x(c, m)] = (_Float16)rpe[(size_t)m * DIM_ + r];   // m = hop index here
  }
  for (int i = tid; i < VJ_ * VJ_; i += 128) {
    sHops[i] = hops[i];
    sOut[i] = outer_w[(size_t)h * VJ_ * VJ_ + i];
  }
  if (tid < HD_) sW1[tid] = w1[h * HD_ + tid];
  if (tid == 0) {
    float a1 = 0.f, a2 = 0.f;
    for (int i = 0; i < HD_ / 2; ++i) { a1 += lq1[i] * lk1[i]; a2 += lq2[i] * lk2[i]; }
    sLam = expf(a1) - expf(a2) + (0.8f - 0.6f * expf(-1.8f));   // LAM_INIT(H=6)
    sAlpha = alpha[0];
  }
  __syncthreads();
  if (tid < VJ_) {   // d[m] = w1 . Ek[m]
    float acc = 0.f;
    for (int c = 0; c < HD_; ++c) acc += sW1[c] * sEk[tid * HD_ + c];
    sD[tid] = acc;
  }
  // phase 1: score WMMAs (one product per wave, 2x2 tiles, K=32)
  {
    const _Float16* Aop = (wave < 2) ? fQ1 : fQ2;
    const _Float16* Bop = (wave == 0) ? fKE1 : (wave == 1) ? fRpe : (wave == 2) ? fKE2 : fRpe;
    float* Cout = (wave == 0) ? sMain1 : (wave == 1) ? sQr1 : (wave == 2) ? sMain2 : sQr2;
#pragma unroll
    for (int rt = 0; rt < 2; ++rt) {
      v16h a = *(const v16h*)(Aop + (rt * 16 + (lane & 15)) * 32 + ((lane & 16) ? 16 : 0));
#pragma unroll
      for (int ct = 0; ct < 2; ++ct) {
        v16h b = *(const v16h*)(Bop + (ct * 32 + lane) * 16);
        v8f c0 = {};
        c0 = wmma16(a, b, c0);
        int mrow = rt * 16 + ((lane & 16) ? 8 : 0);
        int nc2 = ct * 16 + (lane & 15);
#pragma unroll
        for (int r = 0; r < 8; ++r) Cout[(mrow + r) * 32 + nc2] = c0[r];
      }
    }
  }
  __syncthreads();
  // phase 2: gather rel-pos bias, add d, scale
  for (int p = tid; p < VJ_ * VJ_; p += 128) {
    int nj = p / VJ_, m = p - nj * VJ_;
    int hop = sHops[p];
    sMain1[nj * 32 + m] = (sMain1[nj * 32 + m] + sQr1[nj * 32 + hop] + sD[m]) * SCALE_;
    sMain2[nj * 32 + m] = (sMain2[nj * 32 + m] + sQr2[nj * 32 + hop] + sD[m]) * SCALE_;
  }
  __syncthreads();
  // phase 3: softmax rows (stride 32)
  {
    float* S = nullptr; int row = -1;
    if (tid < VJ_)                        { S = sMain1; row = tid; }
    else if (tid >= 32 && tid < 32 + VJ_) { S = sMain2; row = tid - 32; }
    if (S) {
      float mx = -1e30f;
      for (int m = 0; m < VJ_; ++m) mx = fmaxf(mx, S[row * 32 + m]);
      float sum = 0.f;
      for (int m = 0; m < VJ_; ++m) { float e = expf(S[row * 32 + m] - mx); S[row * 32 + m] = e; sum += e; }
      float inv = 1.0f / sum;
      for (int m = 0; m < VJ_; ++m) S[row * 32 + m] *= inv;
    }
  }
  __syncthreads();
  // combine: W = (alpha*a1 + outer) - lam*(alpha*a2 + outer)  -> f16 A-fragment
  for (int p = tid; p < VJ_ * VJ_; p += 128) {
    int nj = p / VJ_, m = p - nj * VJ_;
    float w1v = sAlpha * sMain1[nj * 32 + m] + sOut[p];
    float w2v = sAlpha * sMain2[nj * 32 + m] + sOut[p];
    fWc[frag_off_w(nj, m)] = (_Float16)(w1v - sLam * w2v);
  }
  __syncthreads();
  // phase 4: X = W * Vv (one 16x16 tile per wave, K = 32)
  {
    int rt = wave & 1, ct = wave >> 1;
    v16h a = *(const v16h*)(fWc + (rt * 16 + (lane & 15)) * 32 + ((lane & 16) ? 16 : 0));
    v16h b = *(const v16h*)(fVv + (ct * 32 + lane) * 16);
    v8f c0 = {};
    c0 = wmma16(a, b, c0);
    int c = ct * 16 + (lane & 15);
#pragma unroll
    for (int r = 0; r < 8; ++r) {
      int nj = rt * 16 + ((lane & 16) ? 8 : 0) + r;
      if (nj < VJ_)
        xsp[packed_idx(6, h * HD_ + c, (int)colbase + nj)] = (_Float16)c0[r];
    }
  }
}

// ---------------- grouped head projection (WMMA, K=32): d_out += proj_w[h]*XS + b ----------------
__global__ __launch_bounds__(128) void proj_gemm_kernel(const _Float16* __restrict__ xsp,
                                                        const _Float16* __restrict__ Wp,
                                                        const float* __restrict__ proj_b,
                                                        float* __restrict__ out) {
  __shared__ alignas(32) _Float16 ldsX[2048];
  __shared__ alignas(32) _Float16 ldsW[1024];
  const int tid = threadIdx.x;
  const int wave = tid >> 5, lane = tid & 31;
  const int h = blockIdx.y;
  const int colBase = blockIdx.x * 64;
  const _Float16* xsrc = xsp + ((size_t)blockIdx.x * 6 + h) * 2048;   // XS tile (kt = h)
  const _Float16* wsrc = Wp + (size_t)h * 1024;
#if HAVE_TDM
  if (wave == 0) { tdm_copy_f16(xsrc, ldsX, 2048); __builtin_amdgcn_s_wait_tensorcnt(0); }
  else if (wave == 1) { tdm_copy_f16(wsrc, ldsW, 1024); __builtin_amdgcn_s_wait_tensorcnt(0); }
#else
  for (int e = tid; e < 2048; e += 128) ldsX[e] = xsrc[e];
  for (int e = tid; e < 1024; e += 128) ldsW[e] = wsrc[e];
#endif
  __syncthreads();
  const int mBase = 16 * (wave & 1);
  const int nHalf = (wave >> 1) * 2;
  v16h a = *(const v16h*)(ldsW + (mBase + (lane & 15)) * 32 + ((lane & 16) ? 16 : 0));
  v8f acc[2] = {};
#pragma unroll
  for (int nt = 0; nt < 2; ++nt) {
    v16h b = *(const v16h*)(ldsX + ((nHalf + nt) * 32 + lane) * 16);
    acc[nt] = wmma16(a, b, acc[nt]);
  }
  const int mrow = mBase + ((lane & 16) ? 8 : 0);
  const int nc = lane & 15;
#pragma unroll
  for (int nt = 0; nt < 2; ++nt) {
    int col = colBase + (nHalf + nt) * 16 + nc;
    int n = col / TVX, tv = col - n * TVX;
#pragma unroll
    for (int r = 0; r < 8; ++r) {
      int m = mrow + r;
      float* o = &out[((size_t)n * DIM_ + h * HD_ + m) * TVX + tv];
      *o += acc[nt][r] + proj_b[h * HD_ + m];
    }
  }
}

// ---------------- host side ----------------
extern "C" void kernel_launch(void* const* d_in, const int* in_sizes, int n_in,
                              void* d_out, int out_size, void* d_ws, size_t ws_size,
                              hipStream_t stream) {
  (void)in_sizes; (void)n_in; (void)out_size; (void)ws_size;
  const float* x      = (const float*)d_in[0];
  const int*   jl     = (const int*)d_in[1];
  const float* A      = (const float*)d_in[3];
  const int*   hops   = (const int*)d_in[4];
  const float* norm_g = (const float*)d_in[5];
  const float* norm_b = (const float*)d_in[6];
  const float* skip_w = (const float*)d_in[7];
  const float* pe_w   = (const float*)d_in[8];
  const float* rpe    = (const float*)d_in[9];
  const float* w1     = (const float*)d_in[10];
  const float* outer  = (const float*)d_in[11];
  const float* alpha  = (const float*)d_in[12];
  const float* v_w    = (const float*)d_in[13];
  const float* k_w    = (const float*)d_in[14];
  const float* q_w    = (const float*)d_in[15];
  const float* proj_w = (const float*)d_in[16];
  const float* proj_b = (const float*)d_in[17];
  const float* lq1    = (const float*)d_in[18];
  const float* lk1    = (const float*)d_in[19];
  const float* lq2    = (const float*)d_in[20];
  const float* lk2    = (const float*)d_in[21];
  const float* Mm     = (const float*)d_in[22];
  const float* gcn_w  = (const float*)d_in[23];
  const float* gcn_b  = (const float*)d_in[24];
  const float* bn_g   = (const float*)d_in[25];
  const float* bn_b   = (const float*)d_in[26];
  float* out = (float*)d_out;

  char* p = (char*)d_ws;
  auto alloc = [&p](size_t bytes) -> void* {
    void* r = (void*)p;
    p += (bytes + 255) & ~(size_t)255;
    return r;
  };
  _Float16* XNp  = (_Float16*)alloc((size_t)CIN_ * NCOLS * 2);
  _Float16* Zp   = (_Float16*)alloc((size_t)CIN_ * ECOLS * 2);
  float*    Ebuf = (float*)   alloc((size_t)DIM_ * ECOLS * 4);
  _Float16* VKQ  = (_Float16*)alloc((size_t)960 * NCOLS * 2);
  _Float16* XSp  = (_Float16*)alloc((size_t)DIM_ * NCOLS * 2);
  _Float16* AGGp = (_Float16*)alloc((size_t)3 * CIN_ * NCOLS * 2);
  _Float16* skipWp = (_Float16*)alloc((size_t)DIM_ * CIN_ * 2);
  _Float16* peWp   = (_Float16*)alloc((size_t)DIM_ * CIN_ * 2);
  _Float16* vkqWp  = (_Float16*)alloc((size_t)960 * CIN_ * 2);
  _Float16* gcnWp  = (_Float16*)alloc((size_t)DIM_ * 3 * CIN_ * 2);
  _Float16* projWp = (_Float16*)alloc((size_t)HH_ * HD_ * HD_ * 2);

  // 0. pack weights to fragment-order f16 (tiny, reused by 1500 column blocks)
  pack_w_kernel<<<dim3((192 * 96 + 255) / 256), 256, 0, stream>>>(skip_w, skipWp, 192, 96, 0);
  pack_w_kernel<<<dim3((192 * 96 + 255) / 256), 256, 0, stream>>>(pe_w, peWp, 192, 96, 0);
  pack_w_kernel<<<dim3((192 * 96 + 255) / 256), 256, 0, stream>>>(v_w, vkqWp, 192, 96, 0);
  pack_w_kernel<<<dim3((384 * 96 + 255) / 256), 256, 0, stream>>>(k_w, vkqWp, 384, 96, 192);
  pack_w_kernel<<<dim3((384 * 96 + 255) / 256), 256, 0, stream>>>(q_w, vkqWp, 384, 96, 576);
  pack_w_gcn_kernel<<<dim3((192 * 288 + 255) / 256), 256, 0, stream>>>(gcn_w, gcnWp);
  pack_w_proj_kernel<<<dim3((6144 + 255) / 256), 256, 0, stream>>>(proj_w, projWp);
  // 1. LayerNorm and group pooling (write packed tiles)
  layernorm_kernel<<<dim3(NCOLS / 128), 128, 0, stream>>>(x, norm_g, norm_b, XNp);
  zpool_kernel<<<dim3((CIN_ * NB_ * TT_ + 127) / 128), 128, 0, stream>>>(x, jl, Zp);
  // 2. skip GEMM (writes all of d_out)
  gemm_wmma<0, 2><<<dim3(NCOLS / 64, DIM_ / 64), 128, 0, stream>>>(
      x, skipWp, out, CIN_, NCOLS, nullptr, nullptr, nullptr);
  // 3. positional-embedding GEMM
  gemm_wmma<1, 0><<<dim3(ECOLS / 64, DIM_ / 64), 128, 0, stream>>>(
      Zp, peWp, Ebuf, CIN_, ECOLS, nullptr, nullptr, nullptr);
  // 4. fused v/k/q GEMM (960 output rows)
  gemm_wmma<1, 1><<<dim3(NCOLS / 64, 960 / 64), 128, 0, stream>>>(
      XNp, vkqWp, VKQ, CIN_, NCOLS, nullptr, nullptr, nullptr);
  // 5. GCN aggregation (WMMA)
  agg_kernel<<<dim3(NB_ * TT_), 128, 0, stream>>>(x, A, Mm, AGGp);
  // 6. attention (WMMA scores + WMMA output)
  attn_kernel<<<dim3(NB_ * TT_, HH_), 128, 0, stream>>>(
      VKQ, Ebuf, rpe, w1, outer, hops, jl, alpha, lq1, lk1, lq2, lk2, XSp);
  // 7. grouped projection (+= into d_out)
  proj_gemm_kernel<<<dim3(NCOLS / 64, HH_), 128, 0, stream>>>(XSp, projWp, proj_b, out);
  // 8. GCN GEMM with bias/BN/ReLU epilogue (+= into d_out)
  gemm_wmma<1, 3><<<dim3(NCOLS / 64, DIM_ / 64), 128, 0, stream>>>(
      AGGp, gcnWp, out, 3 * CIN_, NCOLS, gcn_b, bn_g, bn_b);
}
